// UnitaryExpert_26749056319683
// MI455X (gfx1250) — matrix operand order repaired and checked by reading, not compiled, not run on hardware
//
#include <hip/hip_runtime.h>
#include <math.h>

// out = x @ (M1 @ G @ M2)   where for W in {u1,u2}:
//   B  = 0.5*(W^T - W)            (= A^T; A skew => A^T = -A)
//   M  = (I+B) * sum_{k=0..5} B^k = (I+B)^2 * (I + B^2 + B^4)
//   G  = block-diag 2x2 rotation by theta (quantum_activation is linear!)
// Whole network = ONE 16384x2048x2048 GEMM + 7 small 2048^3 GEMMs.
// GEMM: f16-input / f32-accumulate WMMA (v_wmma_f32_16x16x32_f16),
// f32->f16 conversion while staging to LDS, double-buffered LDS with
// register prefetch (1 barrier / K-step), compile-time N=K for immediate
// address offsets, fused epilogues (I + aux add; G-rotation; NT stores).
// Big GEMM uses a 256x128 block (64x64 per wave, 16 WMMA tiles) to halve
// LDS-issue and barrier cost per WMMA; small GEMMs keep 128x128 blocks so
// the 2048^2 grids still fill the machine (256 workgroups).

typedef __attribute__((ext_vector_type(16))) _Float16 v16h;
typedef __attribute__((ext_vector_type(2)))  _Float16 h2;
typedef __attribute__((ext_vector_type(8)))  float    v8f;

#define BN 128
#define BK 32
#define LSTR 34   // halves per LDS row (32 + 2 pad -> 68B rows, 17-bank step)

// MODE bits: 1 = epilogue C = I + A*B + aux ; 2 = B-rows pre-rotated by G ;
//            4 = nontemporal C stores (streaming output)
// TM: block rows (128 or 256). Block cols fixed at 128. 8 waves of
// (TM/4) x 64 each; MT = TM/64 m-tiles, 4 n-tiles per wave.
template<int N, int K, int MODE, int TM>
__global__ __launch_bounds__(256) void gemm_f16wmma(
    const float* __restrict__ A, const float* __restrict__ B,
    float* __restrict__ C, int M,
    const float* __restrict__ aux, const float* __restrict__ theta)
{
    constexpr int MT  = TM / 64;     // m-tiles per wave (2 or 4)
    constexpr int AIT = TM / 32;     // A staging float4s per thread (4 or 8)

    __shared__ _Float16 As[2][TM][LSTR];    // [buf][row][k]
    __shared__ _Float16 Bs[2][BN][LSTR];    // [buf][col][k] (N-major: K contiguous)

    const int tid   = threadIdx.x;
    const int lane  = tid & 31;
    const int wid   = tid >> 5;          // 0..7
    const int waveM = wid >> 1;          // 0..3 -> (TM/4)-row slab
    const int waveN = wid & 1;           // 0..1 -> 64-col slab
    const int rowBase = blockIdx.y * TM;
    const int colBase = blockIdx.x * BN;
    const int lhalf = lane >> 4;
    const int l15   = lane & 15;

    // global staging indices
    const int ar  = tid >> 3;            // 0..31 (+ it*32)
    const int akv = (tid & 7) * 4;       // 0..28
    const int bkr = tid >> 5;            // 0..7  (+ it*8)
    const int bnv = (tid & 31) * 4;      // 0..124

    const float* aBase = A + (size_t)(rowBase + ar) * K + akv;
    const float* bBase = B + (size_t)bkr * N + colBase + bnv;

    // G-rotation setup (MODE&2): row r of (G*B): even r -> c*B[r]+s*B[r+1],
    // odd r -> c*B[r]-s*B[r-1]. Parity of global row == parity of bkr.
    float cth = 1.0f, sgn = 0.0f;
    int bPart = 0;
    if (MODE & 2) {
        float th = theta[0];
        cth = cosf(th);
        float sth = sinf(th);
        sgn   = (bkr & 1) ? -sth : sth;
        bPart = (bkr & 1) ? -N : N;
    }

    float4 aR[AIT], bR[4];

    auto gload = [&](int k0) {
        #pragma unroll
        for (int it = 0; it < AIT; ++it)
            aR[it] = *(const float4*)(aBase + (size_t)k0 + it * (32 * K));
        #pragma unroll
        for (int it = 0; it < 4; ++it) {
            const float* p = bBase + (size_t)k0 * N + it * (8 * N);
            bR[it] = *(const float4*)(p);
            if (MODE & 2) {
                const float4 vp = *(const float4*)(p + bPart);
                bR[it].x = cth * bR[it].x + sgn * vp.x;
                bR[it].y = cth * bR[it].y + sgn * vp.y;
                bR[it].z = cth * bR[it].z + sgn * vp.z;
                bR[it].w = cth * bR[it].w + sgn * vp.w;
            }
        }
    };
    auto sstore = [&](int buf) {
        #pragma unroll
        for (int it = 0; it < AIT; ++it) {
            // 4 contiguous halves, 4B-aligned -> two packed 32-bit DS stores
            h2* pa = (h2*)&As[buf][ar + it * 32][akv];
            pa[0] = h2{(_Float16)aR[it].x, (_Float16)aR[it].y};
            pa[1] = h2{(_Float16)aR[it].z, (_Float16)aR[it].w};
        }
        #pragma unroll
        for (int it = 0; it < 4; ++it) {
            Bs[buf][bnv + 0][bkr + it * 8] = (_Float16)bR[it].x;
            Bs[buf][bnv + 1][bkr + it * 8] = (_Float16)bR[it].y;
            Bs[buf][bnv + 2][bkr + it * 8] = (_Float16)bR[it].z;
            Bs[buf][bnv + 3][bkr + it * 8] = (_Float16)bR[it].w;
        }
    };

    v8f acc[MT][4] = {};
    union Frag { v16h v; unsigned int u[8]; };

    gload(0);
    sstore(0);
    __syncthreads();

    const int nIter = K / BK;
    for (int i = 0; i < nIter; ++i) {
        const int cur = i & 1;
        if (i + 1 < nIter) gload((i + 1) * BK);

        // B 32x16 f16 frag: lane -> col N=(l&15); VGPR v: K = 16*lhalf + 2v,2v+1
        Frag bf[4];
        #pragma unroll
        for (int nt = 0; nt < 4; ++nt) {
            int ncol = waveN * 64 + nt * 16 + l15;
            const unsigned int* p = (const unsigned int*)&Bs[cur][ncol][lhalf * 16];
            #pragma unroll
            for (int j = 0; j < 8; ++j) bf[nt].u[j] = p[j];
        }
        // A 16x32 f16 frag: lane -> row M=(l&15); K = 16*lhalf + 0..15
        Frag af[MT];
        #pragma unroll
        for (int mt = 0; mt < MT; ++mt) {
            int r = waveM * (TM / 4) + mt * 16 + l15;
            const unsigned int* p0 = (const unsigned int*)&As[cur][r][lhalf * 16];
            const unsigned int* p1 = (const unsigned int*)&As[cur][r][lhalf * 16 + 8];
            #pragma unroll
            for (int j = 0; j < 4; ++j) { af[mt].u[j] = p0[j]; af[mt].u[4 + j] = p1[j]; }
        }

        #pragma unroll
        for (int mt = 0; mt < MT; ++mt)
            #pragma unroll
            for (int nt = 0; nt < 4; ++nt)
                acc[mt][nt] = __builtin_amdgcn_wmma_f32_16x16x32_f16(
                    false, af[mt].v, false, bf[nt].v,
                    (short)0, acc[mt][nt], false, false);

        if (i + 1 < nIter) sstore(cur ^ 1);
        __syncthreads();
    }

    // Epilogue. C/D f32 layout: VGPR r -> row r + 8*lhalf, col = lane&15.
    #pragma unroll
    for (int mt = 0; mt < MT; ++mt) {
        const int tr = rowBase + waveM * (TM / 4) + mt * 16 + lhalf * 8;
        #pragma unroll
        for (int nt = 0; nt < 4; ++nt) {
            const int tc = colBase + waveN * 64 + nt * 16 + l15;
            #pragma unroll
            for (int r = 0; r < 8; ++r) {
                const int row = tr + r;
                float v = acc[mt][nt][r];
                if (MODE & 1)
                    v += ((row == tc) ? 1.0f : 0.0f) + aux[(size_t)row * N + tc];
                if (MODE & 4)
                    __builtin_nontemporal_store(v, &C[(size_t)row * N + tc]);
                else
                    C[(size_t)row * N + tc] = v;
            }
        }
    }
}

// S = 0.5*(W^T - W)
__global__ void skew_kernel(const float* __restrict__ W, float* __restrict__ S, int n) {
    int idx = blockIdx.x * blockDim.x + threadIdx.x;
    if (idx >= n * n) return;
    int i = idx / n, j = idx - i * n;
    S[idx] = 0.5f * (W[(size_t)j * n + i] - W[idx]);
}

// E = I + 2*S + C, in place on S
__global__ void polyE_kernel(float* __restrict__ S, const float* __restrict__ C, int n) {
    int idx = blockIdx.x * blockDim.x + threadIdx.x;
    if (idx >= n * n) return;
    int i = idx / n, j = idx - i * n;
    S[idx] = ((i == j) ? 1.0f : 0.0f) + 2.0f * S[idx] + C[idx];
}

#define DM 2048   // model dim (fixed by the reference problem)

extern "C" void kernel_launch(void* const* d_in, const int* in_sizes, int n_in,
                              void* d_out, int out_size, void* d_ws, size_t ws_size,
                              hipStream_t stream) {
    const float* x     = (const float*)d_in[0];
    const float* u1    = (const float*)d_in[1];
    const float* u2    = (const float*)d_in[2];
    const float* theta = (const float*)d_in[3];

    const int n = DM;
    const int Mrows = in_sizes[0] / n;          // 16384
    const size_t nn = (size_t)n * n;

    float* F0 = (float*)d_ws;
    float* F1 = F0 + nn;
    float* F2 = F1 + nn;
    float* F3 = F2 + nn;

    const dim3 blkE(256);
    const dim3 grdE((unsigned)((nn + 255) / 256));
    const dim3 blkG(256);
    const dim3 grdS(n / BN, n / 128);           // small GEMMs: 128x128 blocks
    const dim3 grdB(n / BN, Mrows / 256);       // big GEMM: 256x128 blocks

    // ---- M1 = (I+B1)^2 (I + B1^2 + B1^4) ----
    skew_kernel<<<grdE, blkE, 0, stream>>>(u1, F0, n);                                      // F0 = B1
    gemm_f16wmma<DM,DM,0,128><<<grdS, blkG, 0, stream>>>(F0, F0, F1, n, nullptr, nullptr);  // F1 = C1 = B1^2
    gemm_f16wmma<DM,DM,1,128><<<grdS, blkG, 0, stream>>>(F1, F1, F2, n, F1, nullptr);       // F2 = D1 = I + C1 + C1^2
    polyE_kernel<<<grdE, blkE, 0, stream>>>(F0, F1, n);                                     // F0 = E1 = I + 2B1 + C1
    gemm_f16wmma<DM,DM,0,128><<<grdS, blkG, 0, stream>>>(F0, F2, F1, n, nullptr, nullptr);  // F1 = M1

    // ---- M2 = (I+B2)^2 (I + B2^2 + B2^4) ----
    skew_kernel<<<grdE, blkE, 0, stream>>>(u2, F0, n);                                      // F0 = B2
    gemm_f16wmma<DM,DM,0,128><<<grdS, blkG, 0, stream>>>(F0, F0, F2, n, nullptr, nullptr);  // F2 = C2
    gemm_f16wmma<DM,DM,1,128><<<grdS, blkG, 0, stream>>>(F2, F2, F3, n, F2, nullptr);       // F3 = D2 = I + C2 + C2^2
    polyE_kernel<<<grdE, blkE, 0, stream>>>(F0, F2, n);                                     // F0 = E2
    gemm_f16wmma<DM,DM,0,128><<<grdS, blkG, 0, stream>>>(F0, F3, F2, n, nullptr, nullptr);  // F2 = M2

    // ---- M_total = M1 * (G * M2) ; out = x * M_total ----
    gemm_f16wmma<DM,DM,2,128><<<grdS, blkG, 0, stream>>>(F1, F2, F3, n, nullptr, theta);    // F3 = M_total
    gemm_f16wmma<DM,DM,4,256><<<grdB, blkG, 0, stream>>>(x, F3, (float*)d_out, Mrows,
                                                         nullptr, nullptr);                 // out
}
